// VectorQuantizer_1846835937346
// MI455X (gfx1250) — matrix operand order repaired and checked by reading, not compiled, not run on hardware
//
#include <hip/hip_runtime.h>
#include <stdint.h>
#include <stddef.h>

typedef __attribute__((ext_vector_type(16))) __bf16 v16bf;
typedef __attribute__((ext_vector_type(8)))  __bf16 v8bf;
typedef __attribute__((ext_vector_type(8)))  float  v8f;

namespace {
constexpr int kTok        = 32768;   // 32*32*32 tokens
constexpr int kDim        = 512;     // embedding dim (GEMM K)
constexpr int kCode       = 512;     // codebook entries (GEMM N)
constexpr int kRowsPerBlk = 128;     // tokens per workgroup (8 waves x 16 rows)
constexpr int kThreads    = 256;
constexpr int kIters      = 16;      // outer iterations, 32 codes each
constexpr int kDSteps     = 16;      // 512 / 32

// LDS byte offsets -- fragment-major layouts, no padding needed.
// X hi/lo: 128*512 bf16 each = 131072 B each.
// E tile hi/lo: 32*512 bf16 each = 32768 B each.  Total = exactly 320 KB.
constexpr int kXhOff = 0;
constexpr int kXlOff = 131072;
constexpr int kEhOff = 262144;
constexpr int kElOff = 294912;
constexpr int kSmemBytes = 327680;
} // namespace

// ---------------------------------------------------------------------------
// Prep kernel 1: split E fp32 -> bf16 hi/lo, written in B-FRAGMENT-MAJOR order:
// chunk c (16 elems, 32B) = (((nt*16 + dsb)*2 + kg)*16 + n), covering
// code k = nt*16+n, dims d = dsb*32 + kg*16 + [0..15].  One chunk per thread.
// ---------------------------------------------------------------------------
__global__ __launch_bounds__(kThreads)
void vq_convert_e(const float* __restrict__ E,
                  __bf16* __restrict__ ehG, __bf16* __restrict__ elG) {
  const int c   = blockIdx.x * kThreads + threadIdx.x;  // 0..16383
  const int n   = c & 15;
  const int kg  = (c >> 4) & 1;
  const int dsb = (c >> 5) & 15;
  const int nt  = c >> 9;
  const int k   = nt * 16 + n;
  const int d0  = dsb * 32 + kg * 16;
  const float4* g = (const float4*)(E + k * kDim + d0);
  float f[16];
#pragma unroll
  for (int q = 0; q < 4; ++q) {
    const float4 v = g[q];
    f[q * 4 + 0] = v.x; f[q * 4 + 1] = v.y; f[q * 4 + 2] = v.z; f[q * 4 + 3] = v.w;
  }
  v8bf h0, h1, l0, l1;
#pragma unroll
  for (int j = 0; j < 8; ++j) {
    __bf16 hb = (__bf16)f[j];
    h0[j] = hb; l0[j] = (__bf16)(f[j] - (float)hb);
    hb = (__bf16)f[j + 8];
    h1[j] = hb; l1[j] = (__bf16)(f[j + 8] - (float)hb);
  }
  *(v8bf*)(ehG + c * 16)     = h0;
  *(v8bf*)(ehG + c * 16 + 8) = h1;
  *(v8bf*)(elG + c * 16)     = l0;
  *(v8bf*)(elG + c * 16 + 8) = l1;
}

// ---------------------------------------------------------------------------
// Prep kernel 2: bias[c] = sum_j E[j][c]^2  (the reference's axis=0 quirk).
// ---------------------------------------------------------------------------
__global__ __launch_bounds__(kThreads)
void vq_bias(const float* __restrict__ E, float* __restrict__ bias) {
  const int c = blockIdx.x * kThreads + threadIdx.x;   // 0..511
  float s = 0.f;
  for (int j = 0; j < kCode; ++j) {
    const float v = E[j * kDim + c];
    s = fmaf(v, v, s);
  }
  bias[c] = s;
}

// ---------------------------------------------------------------------------
// Main kernel: per wave 16 rows x 32 codes per pass (A reused across 2 N-tiles
// -> 2.0 b128 loads per WMMA), fragment-major conflict-free LDS, E tiles
// software-pipelined through registers, 3-term split-bf16 for fp32-grade argmin.
// ---------------------------------------------------------------------------
__global__ __launch_bounds__(kThreads, 1)
void vq_main(const float* __restrict__ x,
             const float* __restrict__ emb,
             const __bf16* __restrict__ ehG,
             const __bf16* __restrict__ elG,
             const float* __restrict__ biasG,
             float* __restrict__ out) {
  extern __shared__ __align__(32) char smem[];
  char* xhB = smem + kXhOff;
  char* xlB = smem + kXlOff;
  char* ehB = smem + kEhOff;
  char* elB = smem + kElOff;

  const int t   = threadIdx.x;
  const int blk = blockIdx.x;

  // ---- Phase 1: X -> LDS, split to bf16 hi/lo in A-FRAGMENT-MAJOR order ----
  // chunk c (8 elems,16B) = (((s*16 + dsb)*4 + sc)*16 + m): row s*16+m,
  // dims d = dsb*32 + sc*8 + [0..7].  Consecutive threads -> consecutive
  // chunks -> conflict-free LDS stores.
  {
    const float* Xb = x + (size_t)blk * (kRowsPerBlk * kDim);
#pragma unroll 4
    for (int i = 0; i < 32; ++i) {
      const int c   = i * kThreads + t;     // 0..8191
      const int m   = c & 15;
      const int sc  = (c >> 4) & 3;
      const int dsb = (c >> 6) & 15;
      const int s   = c >> 10;
      const int row = s * 16 + m;
      const int d0  = dsb * 32 + sc * 8;
      const float4* g = (const float4*)(Xb + row * kDim + d0);
      const float4 v0 = g[0];
      const float4 v1 = g[1];
      float f[8] = {v0.x, v0.y, v0.z, v0.w, v1.x, v1.y, v1.z, v1.w};
      v8bf h, l;
#pragma unroll
      for (int j = 0; j < 8; ++j) {
        __bf16 hb = (__bf16)f[j];
        h[j] = hb;
        l[j] = (__bf16)(f[j] - (float)hb);
      }
      *(v8bf*)(xhB + c * 16) = h;
      *(v8bf*)(xlB + c * 16) = l;
    }
  }

  const int wave = t >> 5;
  const int lane = t & 31;
  const int mrow = lane & 15;
  const int kg   = lane >> 4;

  // per-lane fragment base pointers (byte offsets per fragment-major layouts)
  const char* pAh = xhB + (size_t)(wave * 16) * 1024 + kg * 512 + mrow * 16;
  const char* pAl = xlB + (size_t)(wave * 16) * 1024 + kg * 512 + mrow * 16;
  const char* pBh = ehB + kg * 512 + mrow * 32;
  const char* pBl = elB + kg * 512 + mrow * 32;

  float minv[8];
  int   mini[8];
#pragma unroll
  for (int r = 0; r < 8; ++r) { minv[r] = 3.402823466e38f; mini[r] = 0; }

  // ---- register-staged prefetch of E tile 0 (32 codes, hi+lo = 64 KB) ------
  uint4 rh[8], rl[8];
  {
    const uint4* ge = (const uint4*)ehG;
    const uint4* gl = (const uint4*)elG;
#pragma unroll
    for (int i = 0; i < 8; ++i) {
      rh[i] = ge[i * kThreads + t];
      rl[i] = gl[i * kThreads + t];
    }
  }
  __syncthreads();   // phase-1 X stores visible

#pragma unroll 1
  for (int it = 0; it < kIters; ++it) {
    // drain staged registers into the LDS tile (linear, conflict-free)
#pragma unroll
    for (int i = 0; i < 8; ++i) {
      *(uint4*)(ehB + (i * kThreads + t) * 16) = rh[i];
      *(uint4*)(elB + (i * kThreads + t) * 16) = rl[i];
    }
    __syncthreads();

    // issue global loads for the NEXT tile; latency hides under compute
    if (it + 1 < kIters) {
      const uint4* ge = (const uint4*)((const char*)ehG + (it + 1) * 32768);
      const uint4* gl = (const uint4*)((const char*)elG + (it + 1) * 32768);
#pragma unroll
      for (int i = 0; i < 8; ++i) {
        rh[i] = ge[i * kThreads + t];
        rl[i] = gl[i * kThreads + t];
      }
    }

    // four independent accumulator chains (2 N-tiles x {hh, cross})
    v8f c00 = {0.f, 0.f, 0.f, 0.f, 0.f, 0.f, 0.f, 0.f};
    v8f c01 = {0.f, 0.f, 0.f, 0.f, 0.f, 0.f, 0.f, 0.f};
    v8f c10 = {0.f, 0.f, 0.f, 0.f, 0.f, 0.f, 0.f, 0.f};
    v8f c11 = {0.f, 0.f, 0.f, 0.f, 0.f, 0.f, 0.f, 0.f};
#pragma unroll
    for (int dsb = 0; dsb < kDSteps; ++dsb) {
      const int aOff = dsb * 1024;
      v8bf a0h = *(const v8bf*)(pAh + aOff);
      v8bf a1h = *(const v8bf*)(pAh + aOff + 256);
      v8bf a0l = *(const v8bf*)(pAl + aOff);
      v8bf a1l = *(const v8bf*)(pAl + aOff + 256);
      v16bf ah, al;
#pragma unroll
      for (int e = 0; e < 8; ++e) {
        ah[e] = a0h[e]; ah[e + 8] = a1h[e];
        al[e] = a0l[e]; al[e + 8] = a1l[e];
      }
      const v16bf bh0 = *(const v16bf*)(pBh + dsb * 1024);
      const v16bf bl0 = *(const v16bf*)(pBl + dsb * 1024);
      const v16bf bh1 = *(const v16bf*)(pBh + (16 + dsb) * 1024);
      const v16bf bl1 = *(const v16bf*)(pBl + (16 + dsb) * 1024);
      c00 = __builtin_amdgcn_wmma_f32_16x16x32_bf16(false, ah, false, bh0, (short)0, c00, false, false);
      c01 = __builtin_amdgcn_wmma_f32_16x16x32_bf16(false, ah, false, bl0, (short)0, c01, false, false);
      c01 = __builtin_amdgcn_wmma_f32_16x16x32_bf16(false, al, false, bh0, (short)0, c01, false, false);
      c10 = __builtin_amdgcn_wmma_f32_16x16x32_bf16(false, ah, false, bh1, (short)0, c10, false, false);
      c11 = __builtin_amdgcn_wmma_f32_16x16x32_bf16(false, ah, false, bl1, (short)0, c11, false, false);
      c11 = __builtin_amdgcn_wmma_f32_16x16x32_bf16(false, al, false, bh1, (short)0, c11, false, false);
    }

    // distance (row-constant ||x||^2 dropped): bias[code] - 2*sim
    const int   code0 = it * 32 + mrow;        // first N-tile's column
    const int   code1 = code0 + 16;            // second N-tile's column
    const float b0 = biasG[code0];
    const float b1 = biasG[code1];
#pragma unroll
    for (int r = 0; r < 8; ++r) {              // C[r] = sim[M = r + 8*kg][code]
      const float d0 = b0 - 2.0f * (c00[r] + c01[r]);
      if (d0 < minv[r]) { minv[r] = d0; mini[r] = code0; }
      const float d1 = b1 - 2.0f * (c10[r] + c11[r]);
      if (d1 < minv[r]) { minv[r] = d1; mini[r] = code1; }
    }
    __syncthreads();  // all waves done reading tile before next overwrite
  }

  // ---- argmin reduction across the 16 lanes that share each row ------------
#pragma unroll
  for (int r = 0; r < 8; ++r) {
    float mv = minv[r];
    int   mi = mini[r];
#pragma unroll
    for (int off = 8; off >= 1; off >>= 1) {   // stays within 16-lane halves
      const float ov = __shfl_xor(mv, off, 32);
      const int   oi = __shfl_xor(mi, off, 32);
      if (ov < mv || (ov == mv && oi < mi)) { mv = ov; mi = oi; }
    }
    minv[r] = mv; mini[r] = mi;                // all lanes in half hold result
  }

  // ---- gather codebook rows, emit STE value out = x + (q - x) --------------
  {
    const float4* E4 = (const float4*)emb;
    const float4* X4 = (const float4*)x + (size_t)blk * (kRowsPerBlk * kDim / 4);
    float4*       O4 = (float4*)out      + (size_t)blk * (kRowsPerBlk * kDim / 4);
#pragma unroll
    for (int j = 0; j < 16; ++j) {             // fully unrolled: mini[j&7] const
      const int idx = __shfl(mini[j & 7], (j >> 3) << 4, 32);
      const int row = wave * 16 + j;
#pragma unroll
      for (int q = 0; q < 4; ++q) {
        const int col = q * 32 + lane;
        const float4 e4 = E4[idx * (kDim / 4) + col];
        const float4 x4 = X4[row * (kDim / 4) + col];
        float4 o;
        o.x = x4.x + (e4.x - x4.x);
        o.y = x4.y + (e4.y - x4.y);
        o.z = x4.z + (e4.z - x4.z);
        o.w = x4.w + (e4.w - x4.w);
        O4[row * (kDim / 4) + col] = o;
      }
    }
  }
}

// ---------------------------------------------------------------------------
extern "C" void kernel_launch(void* const* d_in, const int* in_sizes, int n_in,
                              void* d_out, int out_size, void* d_ws, size_t ws_size,
                              hipStream_t stream) {
  const float* x   = (const float*)d_in[0];   // [32,32,32,512] fp32
  const float* emb = (const float*)d_in[1];   // [512,512] fp32
  float* out = (float*)d_out;

  // workspace: bias (2KB, padded to 4KB) | eh bf16 (512KB) | el bf16 (512KB)
  float*  bias = (float*)d_ws;
  __bf16* ehG  = (__bf16*)((char*)d_ws + 4096);
  __bf16* elG  = (__bf16*)((char*)d_ws + 4096 + (size_t)kCode * kDim * 2);

  vq_convert_e<<<(kCode * kDim / 16) / kThreads, kThreads, 0, stream>>>(emb, ehG, elG);
  vq_bias<<<kCode / kThreads, kThreads, 0, stream>>>(emb, bias);
  vq_main<<<kTok / kRowsPerBlk, kThreads, kSmemBytes, stream>>>(x, emb, ehG, elG, bias, out);
}
